// DNCWrapper_11776800325913
// MI455X (gfx1250) — matrix operand order repaired and checked by reading, not compiled, not run on hardware
//
#include <hip/hip_runtime.h>

// ---------------------------------------------------------------------------
// DNC forward for MI455X (gfx1250, wave32, WMMA + TDM).
//   B=256 T=64 IN=512 H=1024  M=16 W=20 R=4 RV=80  NL=2 NHL=2
// LSTM gate GEMMs + output GEMM on v_wmma_f32_16x16x32_bf16; bf16 weights
// packed once per launch (65MB -> L2 resident across 64 timesteps). Weight
// tiles are DMA'd into LDS by the Tensor Data Mover (tensor_load_to_lds with
// LDS pad = 144B row stride), overlapped with WMMA via double buffering and
// fenced with s_wait_tensorcnt + workgroup barrier. Activations (f32) are
// converted to bf16 in-register while staging to LDS. The tiny DNC memory
// module runs one wave32 per batch element.
// ---------------------------------------------------------------------------

typedef unsigned int uint32;
typedef __attribute__((ext_vector_type(16))) __bf16 v16bf;
typedef __attribute__((ext_vector_type(8)))  float  v8f;
typedef __attribute__((ext_vector_type(4)))  unsigned int u32x4;
typedef __attribute__((ext_vector_type(8)))  int i32x8;
typedef __attribute__((ext_vector_type(4)))  int i32x4;

#ifndef __has_builtin
#define __has_builtin(x) 0
#endif
#if __has_builtin(__builtin_amdgcn_tensor_load_to_lds) && \
    __has_builtin(__builtin_amdgcn_s_wait_tensorcnt)
#define USE_TDM 1
#else
#define USE_TDM 0
#endif

#define BB   256
#define TT   64
#define INP  512
#define HH   1024
#define G4H  4096
#define MSL  16
#define WDW  20
#define RHD  4
#define RVD  80
#define MLW  176    // padded mem-linear output width (163 used)

__device__ inline float sigmoidf_(float x){ return 1.f/(1.f+expf(-x)); }
__device__ inline float softplusf_(float x){ return (x>20.f)? x : log1pf(expf(x)); }

__device__ inline uint32 pk2bf(float a, float b){
  union { __bf16 h[2]; uint32 u; } x;
  x.h[0] = (__bf16)a; x.h[1] = (__bf16)b;
  return x.u;
}

// ---------------------------------------------------------------------------
// Weight conversion / packing kernels (run once per launch; deterministic)
// ---------------------------------------------------------------------------
__global__ void cvt_lstm_w(__bf16* __restrict__ dst, const float* __restrict__ Wih,
                           const float* __restrict__ Whh, int Kin, int Kp){
  long i = (long)blockIdx.x*256 + threadIdx.x;
  if (i >= (long)G4H*Kp) return;
  int n = (int)(i / Kp), k = (int)(i % Kp);
  float v = (k < Kin) ? Wih[(long)n*Kin + k]
          : (k < Kin + HH) ? Whh[(long)n*HH + (k - Kin)] : 0.f;
  dst[i] = (__bf16)v;
}

__global__ void cvt_plain_w(__bf16* __restrict__ dst, const float* __restrict__ W,
                            int N, int K, int Kp){
  long i = (long)blockIdx.x*256 + threadIdx.x;
  if (i >= (long)N*Kp) return;
  int n = (int)(i / Kp), k = (int)(i % Kp);
  dst[i] = (__bf16)((k < K) ? W[(long)n*K + k] : 0.f);
}

__global__ void add_bias2(float* __restrict__ d, const float* __restrict__ a,
                          const float* __restrict__ b, int n){
  int i = blockIdx.x*256 + threadIdx.x;
  if (i < n) d[i] = a[i] + b[i];
}

__global__ void zero_f32(float* __restrict__ p, long n){
  for (long i = (long)blockIdx.x*256 + threadIdx.x; i < n; i += (long)gridDim.x*256)
    p[i] = 0.f;
}

__global__ void copy2d(float* __restrict__ dst, int ds,
                       const float* __restrict__ src, int ss, int rows, int cols){
  int i = blockIdx.x*256 + threadIdx.x;
  if (i >= rows*cols) return;
  int r = i / cols, c = i - r*cols;
  dst[(long)r*ds + c] = src[(long)r*ss + c];
}

// ---------------------------------------------------------------------------
// WMMA GEMM:  C[256 x N] = A(f32, lda, padded-K) * Bw(bf16, [N x Kp])^T + bias
// block tile 64x128, 8 waves (2x4), wave tile 32x32 (2x2 wmma 16x16x32),
// K-block 64, LDS double buffered. B tiles arrive via TDM DMA when available.
// ---------------------------------------------------------------------------
#define GTM 64
#define GTN 128
#define GTK 64
#define LDK 72   // bf16 row stride: 144B (=32 dwords data + 4 dwords TDM pad)

union AFrag { v16bf v; uint32 u[8]; };

__global__ __launch_bounds__(256) void gemm_bf16_wmma(
    const float*  __restrict__ A,  int lda,
    const __bf16* __restrict__ Bw, int ldb,
    const float*  __restrict__ bias,
    float* __restrict__ C, int ldc, int ksteps)
{
  __shared__ __bf16 lA[2][GTM][LDK];
  __shared__ __bf16 lB[2][GTN][LDK];

  const int tid  = threadIdx.x;
  const int lane = tid & 31;
  const int wave = tid >> 5;
  const int wm   = wave >> 2;      // 0..1
  const int wn   = wave & 3;       // 0..3
  const int m0   = blockIdx.y * GTM;
  const int n0   = blockIdx.x * GTN;

  v8f acc[2][2] = {};

  uint32 au[8];    // A staging: already packed bf16 (2 per dword)

  auto gloadA = [&](int k0){
    #pragma unroll
    for (int i = 0; i < 4; ++i){
      int r  = (tid >> 4) + 16*i;
      int c4 = (tid & 15) * 4;
      const float4 f = *(const float4*)&A[(long)(m0 + r)*lda + k0 + c4];
      au[2*i]   = pk2bf(f.x, f.y);
      au[2*i+1] = pk2bf(f.z, f.w);
    }
  };
  auto sstoreA = [&](int buf){
    #pragma unroll
    for (int i = 0; i < 4; ++i){
      int r  = (tid >> 4) + 16*i;
      int c4 = (tid & 15) * 4;
      *(uint32*)&lA[buf][r][c4]   = au[2*i];
      *(uint32*)&lA[buf][r][c4+2] = au[2*i+1];
    }
  };

#if USE_TDM
  // ---- Tensor Data Mover: DMA a 128x64 bf16 tile (row stride ldb elements)
  // into lB[buf] with LDS padding 4 dwords every 32 dwords -> 144B row pitch.
  auto stageB_issue = [&](int buf, int k0){
    if (wave == 0){
      unsigned long long ga =
          (unsigned long long)(size_t)&Bw[(size_t)n0*ldb + k0]; // byte address
      uint32 lds = (uint32)(size_t)&lB[buf][0][0];              // LDS byte offset
      u32x4 g0;
      g0[0] = 1u;                                   // count=1 (valid descriptor)
      g0[1] = lds;                                  // lds_addr
      g0[2] = (uint32)ga;                           // global_addr[31:0]
      g0[3] = (uint32)((ga >> 32) & 0x1FFFFFFu)     // global_addr[56:32]
            | (2u << 30);                           // type = 2 ("image")
      i32x8 g1;
      g1[0] = (1 << 16)    // data_size = 2 bytes
            | (1 << 20)    // pad_enable
            | (4 << 22)    // pad_interval: 32 dwords
            | (3 << 25);   // pad_amount:   4 dwords
      g1[1] = (ldb & 0xFFFF) << 16;                 // tensor_dim0[15:0]
      g1[2] = ((ldb >> 16) & 0xFFFF) | (128 << 16); // dim0[31:16] | tensor_dim1 lo
      g1[3] = (64 << 16);                           // tensor_dim1 hi=0 | tile_dim0=64
      g1[4] = 128;                                  // tile_dim1=128, tile_dim2=0
      g1[5] = ldb;                                  // tensor_dim0_stride[31:0]
      g1[6] = 0;                                    // stride hi | dim1_stride lo
      g1[7] = 0;
      i32x4 z4 = {};
#if defined(__clang_major__) && (__clang_major__ >= 23)
      i32x8 z8 = {};
      __builtin_amdgcn_tensor_load_to_lds(g0, g1, z4, z4, z8, 0);
#else
      __builtin_amdgcn_tensor_load_to_lds(g0, g1, z4, z4, 0);
#endif
    }
  };
  auto stageB_commit = [&](int buf){
    (void)buf;
    if (wave == 0) __builtin_amdgcn_s_wait_tensorcnt(0);
  };
#else
  // ---- fallback: vector loads + ds stores
  uint4 br[4];
  auto stageB_issue = [&](int buf, int k0){
    (void)buf;
    #pragma unroll
    for (int i = 0; i < 4; ++i){
      int r  = (tid >> 3) + 32*i;
      int c8 = (tid & 7) * 8;
      br[i] = *(const uint4*)&Bw[(size_t)(n0 + r)*ldb + k0 + c8];
    }
  };
  auto stageB_commit = [&](int buf){
    #pragma unroll
    for (int i = 0; i < 4; ++i){
      int r  = (tid >> 3) + 32*i;
      int c8 = (tid & 7) * 8;
      *(uint4*)&lB[buf][r][c8] = br[i];
    }
  };
#endif

  auto compute = [&](int buf){
    AFrag af[2], bfr[2];
    #pragma unroll
    for (int kk = 0; kk < GTK; kk += 32){
      #pragma unroll
      for (int mt = 0; mt < 2; ++mt){
        int m  = wm*32 + mt*16 + (lane & 15);
        int kh = (lane >> 4) * 8;
        const __bf16* p = &lA[buf][m][kk];
        #pragma unroll
        for (int j = 0; j < 4; ++j){
          af[mt].u[j]   = *(const uint32*)(p + kh + 2*j);        // K 0..15 half
          af[mt].u[4+j] = *(const uint32*)(p + 16 + kh + 2*j);   // K 16..31 half
        }
      }
      #pragma unroll
      for (int nt = 0; nt < 2; ++nt){
        int n  = wn*32 + nt*16 + (lane & 15);
        int kh = (lane >> 4) * 8;
        const __bf16* p = &lB[buf][n][kk];
        #pragma unroll
        for (int j = 0; j < 8; ++j)
          bfr[nt].u[j] = *(const uint32*)(p + kh + 2*j + ((j >> 2) << 3));
      }
      #pragma unroll
      for (int mt = 0; mt < 2; ++mt)
        #pragma unroll
        for (int nt = 0; nt < 2; ++nt)
          acc[mt][nt] = __builtin_amdgcn_wmma_f32_16x16x32_bf16(
              false, af[mt].v, false, bfr[nt].v, (short)0, acc[mt][nt], false, false);
    }
  };

  // prologue: fill stage 0
  stageB_issue(0, 0);
  gloadA(0); sstoreA(0);
  stageB_commit(0);
  __syncthreads();

  int buf = 0;
  for (int ks = 0; ks < ksteps; ++ks){
    if (ks + 1 < ksteps){
      stageB_issue(buf ^ 1, (ks+1)*GTK);   // DMA next weight tile while computing
      gloadA((ks+1)*GTK);
    }
    if (ks + 2 < ksteps)
      __builtin_prefetch(&Bw[(size_t)(n0 + (tid & 127))*ldb + (ks+2)*GTK], 0, 1);
    compute(buf);
    if (ks + 1 < ksteps){
      sstoreA(buf ^ 1);
      stageB_commit(buf ^ 1);
    }
    __syncthreads();
    buf ^= 1;
  }

  // epilogue: C/D layout (ISA 7.12.2): VGPR r, lanes0-15 -> M=r, lanes16-31 -> M=r+8
  #pragma unroll
  for (int mt = 0; mt < 2; ++mt)
    #pragma unroll
    for (int nt = 0; nt < 2; ++nt){
      int col = n0 + wn*32 + nt*16 + (lane & 15);
      float bv = bias ? bias[col] : 0.f;
      int rbase = m0 + wm*32 + mt*16 + (lane >> 4) * 8;
      #pragma unroll
      for (int r = 0; r < 8; ++r)
        C[(size_t)(rbase + r)*ldc + col] = acc[mt][nt][r] + bv;
    }
}

// ---------------------------------------------------------------------------
// LSTM gate pointwise:  g[b, 0:4H] -> c,h ;  optional clipped layer output
// ---------------------------------------------------------------------------
__global__ __launch_bounds__(256) void lstm_pw(const float* __restrict__ g,
                                               float* __restrict__ c,
                                               float* __restrict__ h,
                                               float* __restrict__ outclip,
                                               int writeOut){
  int i = blockIdx.x*256 + threadIdx.x;     // over B*H
  int b = i >> 10, j = i & 1023;
  const float* gb = g + (long)b*G4H;
  float ig = sigmoidf_(gb[j]);
  float fg = sigmoidf_(gb[HH + j]);
  float gg = tanhf    (gb[2*HH + j]);
  float og = sigmoidf_(gb[3*HH + j]);
  float cn = fg * c[i] + ig * gg;
  float hn = og * tanhf(cn);
  c[i] = cn; h[i] = hn;
  if (writeOut) outclip[i] = fminf(fmaxf(hn, -20.f), 20.f);
}

// ---------------------------------------------------------------------------
// Memory-interface linears: out[b][j] = act[b,:] . Wm[j,:] + bm[j], j < 163
// ---------------------------------------------------------------------------
__global__ __launch_bounds__(256) void memlin(const float* __restrict__ act,
                                              const float* __restrict__ Wm,
                                              const float* __restrict__ bm,
                                              float* __restrict__ out){
  int b = blockIdx.x;
  int wave = threadIdx.x >> 5, lane = threadIdx.x & 31;
  const float* a = act + (long)b*HH;
  for (int j = wave; j < 163; j += 8){
    const float* w = Wm + (long)j*HH;
    float s = 0.f;
    for (int k = lane; k < HH; k += 32) s += a[k]*w[k];
    #pragma unroll
    for (int o = 16; o; o >>= 1) s += __shfl_down(s, o, 32);
    if (lane == 0) out[(long)b*MLW + j] = s + bm[j];
  }
}

// ---------------------------------------------------------------------------
// DNC memory step: one wave32 per batch element.
// ml layout: rk[0:80) rs[80:84) wk[84:104) ws[104] ev[105:125) wv[125:145)
//            fg[145:149) ag[149] wg[150] rm[151:163)
// ---------------------------------------------------------------------------
__global__ __launch_bounds__(32) void memory_step(
    const float* __restrict__ mlv,
    float* __restrict__ mem,  float* __restrict__ link, float* __restrict__ prec,
    float* __restrict__ rw,   float* __restrict__ ww,   float* __restrict__ usage,
    float* __restrict__ rvec)
{
  const int b = blockIdx.x, lane = threadIdx.x;
  const float* L = mlv + (long)b*MLW;
  float* Mb = mem  + (long)b*MSL*WDW;
  float* Lb = link + (long)b*MSL*MSL;
  float* Pb = prec + (long)b*MSL;
  float* Rb = rw   + (long)b*RHD*MSL;
  float* Wb = ww   + (long)b*MSL;
  float* Ub = usage+ (long)b*MSL;
  float* RV = rvec + (long)b*RVD;

  __shared__ float s_rk[RHD][WDW], s_wk[WDW], s_ev[WDW], s_wv[WDW];
  __shared__ float s_rs[RHD], s_fg[RHD], s_modes[RHD][3], s_knr[RHD];
  __shared__ float s_scal[4];                // ws, ag, wg, wk-norm
  __shared__ float s_u[MSL], s_alloc[MSL], s_wcw[MSL], s_wwn[MSL];
  __shared__ float s_pold[MSL], s_rwold[RHD][MSL];
  __shared__ float s_sim[RHD][MSL], s_cw[RHD][MSL], s_fw[RHD][MSL], s_bw[RHD][MSL];

  for (int i = lane; i < RHD*WDW; i += 32) s_rk[i/WDW][i%WDW] = tanhf(L[i]);
  for (int i = lane; i < WDW; i += 32){
    s_wk[i] = tanhf(L[84+i]); s_ev[i] = sigmoidf_(L[105+i]); s_wv[i] = tanhf(L[125+i]);
  }
  if (lane < RHD){ s_rs[lane] = softplusf_(L[80+lane]); s_fg[lane] = sigmoidf_(L[145+lane]); }
  if (lane == 0){ s_scal[0]=softplusf_(L[104]); s_scal[1]=sigmoidf_(L[149]); s_scal[2]=sigmoidf_(L[150]); }
  if (lane < RHD){
    float m0=L[151+3*lane], m1=L[152+3*lane], m2=L[153+3*lane];
    float mx=fmaxf(m0,fmaxf(m1,m2));
    float e0=expf(m0-mx), e1=expf(m1-mx), e2=expf(m2-mx), s=e0+e1+e2;
    s_modes[lane][0]=e0/s; s_modes[lane][1]=e1/s; s_modes[lane][2]=e2/s;
  }
  if (lane < MSL) s_pold[lane] = Pb[lane];
  for (int i = lane; i < RHD*MSL; i += 32) s_rwold[i>>4][i&15] = Rb[i];
  __syncthreads();

  if (lane < MSL){
    float u = Ub[lane];
    u = u + (1.f - u) * Wb[lane];
    float psi = 1.f;
    for (int r = 0; r < RHD; ++r) psi *= (1.f - s_fg[r]*s_rwold[r][lane]);
    u *= psi;
    Ub[lane] = u;
    s_u[lane] = 1e-6f + (1.f - 1e-6f) * u;
  }
  if (lane == 0){
    float kn = 0.f;
    for (int w = 0; w < WDW; ++w) kn += s_wk[w]*s_wk[w];
    s_scal[3] = sqrtf(kn) + 1e-6f;
  }
  __syncthreads();

  if (lane < MSL){
    float nrm = 0.f, dot = 0.f;
    for (int w = 0; w < WDW; ++w){ float mv = Mb[lane*WDW + w]; nrm += mv*mv; dot += mv*s_wk[w]; }
    s_wcw[lane] = dot / ((sqrtf(nrm)+1e-6f) * s_scal[3]) * s_scal[0];
  }
  __syncthreads();
  {
    float mx = -1e30f, sum = 0.f;
    for (int m = 0; m < MSL; ++m) mx = fmaxf(mx, s_wcw[m]);
    for (int m = 0; m < MSL; ++m) sum += expf(s_wcw[m]-mx);
    float e = (lane < MSL) ? expf(s_wcw[lane]-mx)/sum : 0.f;
    __syncthreads();
    if (lane < MSL) s_wcw[lane] = e;
  }
  if (lane == 0){
    int phi[MSL]; float uu[MSL];
    for (int i = 0; i < MSL; ++i){ phi[i]=i; uu[i]=s_u[i]; }
    for (int i = 0; i < MSL; ++i)
      for (int j = i+1; j < MSL; ++j)
        if (uu[phi[j]] < uu[phi[i]]){ int t = phi[i]; phi[i] = phi[j]; phi[j] = t; }
    float pr = 1.f;
    for (int j = 0; j < MSL; ++j){ float su = uu[phi[j]]; s_alloc[phi[j]] = (1.f-su)*pr; pr *= su; }
  }
  __syncthreads();

  if (lane < MSL){
    float w = s_scal[2] * (s_scal[1]*s_alloc[lane] + (1.f - s_scal[1])*s_wcw[lane]);
    s_wwn[lane] = w; Wb[lane] = w;
    for (int k = 0; k < WDW; ++k){
      float mv = Mb[lane*WDW + k];
      Mb[lane*WDW + k] = mv * (1.f - w*s_ev[k]) + w*s_wv[k];
    }
  }
  __syncthreads();

  float sww = 0.f;
  for (int m = 0; m < MSL; ++m) sww += s_wwn[m];
  if (lane < MSL){
    int m = lane;
    for (int j = 0; j < MSL; ++j){
      float lv = Lb[m*MSL + j];
      lv = (1.f - s_wwn[m] - s_wwn[j])*lv + s_wwn[m]*s_pold[j];
      Lb[m*MSL + j] = (m == j) ? 0.f : lv;
    }
    Pb[m] = (1.f - sww)*s_pold[m] + s_wwn[m];
  }
  if (lane < RHD){
    float kn = 0.f;
    for (int w = 0; w < WDW; ++w) kn += s_rk[lane][w]*s_rk[lane][w];
    s_knr[lane] = sqrtf(kn) + 1e-6f;
  }
  __syncthreads();

  for (int e = lane; e < RHD*MSL; e += 32){
    int r = e >> 4, m = e & 15;
    float nrm = 0.f, dot = 0.f;
    for (int w = 0; w < WDW; ++w){ float mv = Mb[m*WDW + w]; nrm += mv*mv; dot += mv*s_rk[r][w]; }
    s_sim[r][m] = dot / ((sqrtf(nrm)+1e-6f) * s_knr[r]) * s_rs[r];
  }
  __syncthreads();
  for (int e = lane; e < RHD*MSL; e += 32){
    int r = e >> 4, m = e & 15;
    float mx = -1e30f, sm = 0.f;
    for (int q = 0; q < MSL; ++q) mx = fmaxf(mx, s_sim[r][q]);
    for (int q = 0; q < MSL; ++q) sm += expf(s_sim[r][q]-mx);
    s_cw[r][m] = expf(s_sim[r][m]-mx)/sm;
  }
  __syncthreads();

  for (int e = lane; e < RHD*MSL; e += 32){
    int r = e >> 4, i = e & 15;
    float f = 0.f, bw2 = 0.f;
    for (int j = 0; j < MSL; ++j){
      f   += Lb[i*MSL + j] * s_rwold[r][j];
      bw2 += Lb[j*MSL + i] * s_rwold[r][j];
    }
    s_fw[r][i] = f; s_bw[r][i] = bw2;
  }
  __syncthreads();

  for (int e = lane; e < RHD*MSL; e += 32){
    int r = e >> 4, m = e & 15;
    float v = s_modes[r][0]*s_bw[r][m] + s_modes[r][1]*s_fw[r][m] + s_modes[r][2]*s_cw[r][m];
    Rb[e] = v;
    s_rwold[r][m] = v;
  }
  __syncthreads();

  for (int e = lane; e < RVD; e += 32){
    int r = e / WDW, w = e - r*WDW;
    float a = 0.f;
    for (int m = 0; m < MSL; ++m) a += s_rwold[r][m] * Mb[m*WDW + w];
    RV[e] = a;
  }
}

// ---------------------------------------------------------------------------
// Host orchestration
// ---------------------------------------------------------------------------
extern "C" void kernel_launch(void* const* d_in, const int* in_sizes, int n_in,
                              void* d_out, int out_size, void* d_ws, size_t ws_size,
                              hipStream_t stream)
{
  (void)in_sizes; (void)n_in; (void)out_size; (void)ws_size;

  // input pointer map (JAX pytree: dict keys sorted; tuples in order)
  auto F = [&](int i)->const float* { return (const float*)d_in[i]; };
  const float* X = F(0);
  const int OUTW = 41, OUTB = 42;
  auto rnnIdx = [&](int l, int cell, int which)->int { // 0 Whh 1 Wih 2 bhh 3 bih
    return 43 + 8*l + 4*cell + which;
  };

  char* ws = (char*)d_ws;
  size_t off = 0;
  auto ALLOC = [&](size_t bytes)->char* {
    char* p = ws + off; off = (off + bytes + 255) & ~(size_t)255; return p;
  };

  static const int Kin4[4] = {592, 1024, 1104, 1024};
  static const int Kp4 [4] = {1664, 2048, 2176, 2048};

  __bf16* Wc[4]; float* bc[4];
  for (int c = 0; c < 4; ++c) Wc[c] = (__bf16*)ALLOC((size_t)G4H*Kp4[c]*2);
  for (int c = 0; c < 4; ++c) bc[c] = (float*)ALLOC((size_t)G4H*4);
  __bf16* WoutB = (__bf16*)ALLOC((size_t)INP*1152*2);
  float *Wm[2], *bm[2];
  for (int l = 0; l < 2; ++l){ Wm[l] = (float*)ALLOC((size_t)MLW*HH*4); bm[l] = (float*)ALLOC(MLW*4); }
  float* g = (float*)ALLOC((size_t)BB*G4H*4);
  float* outL[2]; for (int l = 0; l < 2; ++l) outL[l] = (float*)ALLOC((size_t)BB*HH*4);
  float* ml[2];   for (int l = 0; l < 2; ++l) ml[l]   = (float*)ALLOC((size_t)BB*MLW*4);

  size_t z0 = off;
  float* A00  = (float*)ALLOC((size_t)BB*1664*4);
  float* A1b  = (float*)ALLOC((size_t)BB*2048*4);
  float* A01  = (float*)ALLOC((size_t)BB*2176*4);
  float* Aout = (float*)ALLOC((size_t)BB*1152*4);
  float *hS[2][2], *cS[2][2];
  for (int l = 0; l < 2; ++l) for (int i = 0; i < 2; ++i){
    hS[l][i] = (float*)ALLOC((size_t)BB*HH*4);
    cS[l][i] = (float*)ALLOC((size_t)BB*HH*4);
  }
  float *stRv[2], *stMem[2], *stLink[2], *stPrec[2], *stRw[2], *stWw[2], *stUs[2];
  for (int l = 0; l < 2; ++l){
    stRv[l]   = (float*)ALLOC((size_t)BB*RVD*4);
    stMem[l]  = (float*)ALLOC((size_t)BB*MSL*WDW*4);
    stLink[l] = (float*)ALLOC((size_t)BB*MSL*MSL*4);
    stPrec[l] = (float*)ALLOC((size_t)BB*MSL*4);
    stRw[l]   = (float*)ALLOC((size_t)BB*RHD*MSL*4);
    stWw[l]   = (float*)ALLOC((size_t)BB*MSL*4);
    stUs[l]   = (float*)ALLOC((size_t)BB*MSL*4);
  }
  size_t z1 = off;

  for (int c = 0; c < 4; ++c){
    int l = c >> 1, cell = c & 1;
    long tot = (long)G4H*Kp4[c];
    cvt_lstm_w<<<(int)((tot+255)/256), 256, 0, stream>>>(
        Wc[c], F(rnnIdx(l,cell,1)), F(rnnIdx(l,cell,0)), Kin4[c], Kp4[c]);
    add_bias2<<<(G4H+255)/256, 256, 0, stream>>>(bc[c], F(rnnIdx(l,cell,3)), F(rnnIdx(l,cell,2)), G4H);
  }
  {
    long tot = (long)INP*1152;
    cvt_plain_w<<<(int)((tot+255)/256), 256, 0, stream>>>(WoutB, F(OUTW), INP, 1104, 1152);
  }
  auto CPY = [&](float* dst, int ds, const float* src, int ss, int rows, int cols){
    int n = rows*cols;
    copy2d<<<(n+255)/256, 256, 0, stream>>>(dst, ds, src, ss, rows, cols);
  };
  struct MLSpec { int wRel, rows, o; };
  static const MLSpec mspec[10] = {
    {6,80,0},{10,4,80},{14,20,84},{16,1,104},{2,20,105},
    {18,20,125},{4,4,145},{0,1,149},{12,1,150},{8,12,151}};
  for (int l = 0; l < 2; ++l){
    int base = 1 + 20*l;
    for (int s = 0; s < 10; ++s){
      CPY(Wm[l] + (long)mspec[s].o*HH, HH, F(base + mspec[s].wRel),     HH, mspec[s].rows, HH);
      CPY(bm[l] + mspec[s].o,           0, F(base + mspec[s].wRel + 1),  0, 1, mspec[s].rows);
    }
  }
  zero_f32<<<2048, 256, 0, stream>>>((float*)(ws + z0), (long)((z1 - z0)/4));

  auto GEMM = [&](const float* Ap, int lda, const __bf16* Bp, const float* bias,
                  float* Cp, int ldc, int N, int Kp){
    gemm_bf16_wmma<<<dim3(N/GTN, BB/GTM), 256, 0, stream>>>(Ap, lda, Bp, lda, bias, Cp, ldc, Kp/GTK);
  };
  auto PW = [&](float* cp, float* hp, float* op, int wo){
    lstm_pw<<<(BB*HH)/256, 256, 0, stream>>>(g, cp, hp, op, wo);
  };

  for (int t = 0; t < TT; ++t){
    // ===== layer 0 =====
    CPY(A00,        1664, X + (long)t*INP, TT*INP, BB, INP);
    CPY(A00 + 592,  1664, hS[0][0], HH, BB, HH);
    GEMM(A00, 1664, Wc[0], bc[0], g, G4H, G4H, 1664);
    PW(cS[0][0], hS[0][0], nullptr, 0);

    CPY(A1b,         2048, hS[0][0], HH, BB, HH);
    CPY(A1b + 1024,  2048, hS[0][1], HH, BB, HH);
    GEMM(A1b, 2048, Wc[1], bc[1], g, G4H, G4H, 2048);
    PW(cS[0][1], hS[0][1], outL[0], 1);

    memlin<<<BB, 256, 0, stream>>>(outL[0], Wm[0], bm[0], ml[0]);
    memory_step<<<BB, 32, 0, stream>>>(ml[0], stMem[0], stLink[0], stPrec[0],
                                       stRw[0], stWw[0], stUs[0], stRv[0]);

    // ===== layer 1 =====
    CPY(A01,         2176, outL[0], HH,  BB, HH);
    CPY(A01 + 1024,  2176, stRv[0], RVD, BB, RVD);
    CPY(A01 + 1104,  2176, hS[1][0], HH, BB, HH);
    GEMM(A01, 2176, Wc[2], bc[2], g, G4H, G4H, 2176);
    PW(cS[1][0], hS[1][0], nullptr, 0);

    CPY(A1b,         2048, hS[1][0], HH, BB, HH);
    CPY(A1b + 1024,  2048, hS[1][1], HH, BB, HH);
    GEMM(A1b, 2048, Wc[3], bc[3], g, G4H, G4H, 2048);
    PW(cS[1][1], hS[1][1], outL[1], 1);

    memlin<<<BB, 256, 0, stream>>>(outL[1], Wm[1], bm[1], ml[1]);
    memory_step<<<BB, 32, 0, stream>>>(ml[1], stMem[1], stLink[1], stPrec[1],
                                       stRw[1], stWw[1], stUs[1], stRv[1]);

    // ===== output linear =====
    CPY(Aout,        1152, outL[1], HH,  BB, HH);
    CPY(Aout + 1024, 1152, stRv[1], RVD, BB, RVD);
    GEMM(Aout, 1152, WoutB, F(OUTB), (float*)d_out + (long)t*INP, TT*INP, INP, 1152);
  }
}
